// ChromaEncoder_83872121356743
// MI455X (gfx1250) — matrix operand order repaired and checked by reading, not compiled, not run on hardware
//
#include <hip/hip_runtime.h>

#define BB 2048
#define LL 1024
#define CC 12
#define G4 48
#define MH 64

typedef _Float16 f16;
typedef __attribute__((ext_vector_type(16))) _Float16 v16h;
typedef __attribute__((ext_vector_type(8)))  float    v8f;

// fast transcendentals: v_exp_f32 + v_rcp_f32 (avoid IEEE div expansion,
// which costs ~8 dependent VALU ops on the serial LSTM critical path)
__device__ __forceinline__ float fast_tanh(float x) {
  float e = __expf(2.0f * x);
  return 1.0f - 2.0f * __builtin_amdgcn_rcpf(e + 1.0f);
}
__device__ __forceinline__ float fast_sig(float x) {
  return __builtin_amdgcn_rcpf(1.0f + __expf(-x));
}

// ---------------------------------------------------------------------------
// prep: build per-lane WMMA B-operand register images + fused LSTM biases.
//
// B k order used here (contiguous, k = 16*(lane/16) + half): lane<16 holds K=0..15 of column n=lane
// (halfs 0..15 <-> k), lane>=16 holds K=16..31 of column n=lane-16.
//
// LSTM A provides (via the contiguous-load trick below):
//   k 0..15  = features 0..15  ([x_t(12) | h(0..3)])
//   k 16..23 = ZERO in B (duplicate slots)
//   k 24..31 = features 16..23 (h(4..11))
// MLP A provides: k0..7 = features 0..7, k24..28 = features 8..12, rest zero-B.
// ---------------------------------------------------------------------------
__global__ void prep_kernel(const float* __restrict__ Wih0, const float* __restrict__ Whh0,
                            const float* __restrict__ bih0, const float* __restrict__ bhh0,
                            const float* __restrict__ Wih1, const float* __restrict__ Whh1,
                            const float* __restrict__ bih1, const float* __restrict__ bhh1,
                            const float* __restrict__ W1,
                            f16* __restrict__ bimg, float* __restrict__ biases) {
  int tid = threadIdx.x;
  for (int g = tid; g < G4; g += blockDim.x) {
    biases[g]      = bih0[g] + bhh0[g];
    biases[G4 + g] = bih1[g] + bhh1[g];
  }
  // 10 images: 0..2 = LSTM layer0 N-tiles, 3..5 = layer1, 6..9 = W1 N-tiles.
  for (int idx = tid; idx < 10 * 512; idx += blockDim.x) {
    int img = idx >> 9, rem = idx & 511, lane = rem >> 4, h = rem & 15;
    int n = lane & 15;
    bool lo = lane < 16;
    float v = 0.0f;
    if (img < 6) {
      int nt = img % 3;
      const float* Wih = (img < 3) ? Wih0 : Wih1;
      const float* Whh = (img < 3) ? Whh0 : Whh1;
      int g = nt * 16 + n;                     // gate index 0..47
      if (lo) {                                // k = h: features 0..15
        v = (h < 12) ? Wih[g * 12 + h] : Whh[g * 12 + (h - 12)];
      } else {                                 // k = 16+h
        if (h >= 8) v = Whh[g * 12 + (h - 4)]; // k24..31 -> features 16..23
      }
    } else {
      int u = (img - 6) * 16 + n;              // MLP hidden unit 0..63
      if (lo) { if (h < 8) v = W1[u * 13 + h]; }          // k0..7 -> feat 0..7
      else    { if (h >= 8 && h < 13) v = W1[u * 13 + h]; } // k24..28 -> feat 8..12
    }
    bimg[idx] = (f16)v;
  }
}

// ---------------------------------------------------------------------------
// encoder: xf = tanh(x @ W_loc^T + b_loc), written time-major as f16 rows of
// 16 halfs (12 real + 4 pad) so the LSTM can do 128-bit loads.
// ---------------------------------------------------------------------------
__global__ void enc_kernel(const float* __restrict__ x, const float* __restrict__ W_loc,
                           const float* __restrict__ b_loc, f16* __restrict__ xf16) {
  __shared__ float Wl[144];
  __shared__ float bl[12];
  int tid = threadIdx.x;
  if (tid < 144) Wl[tid] = W_loc[tid];
  if (tid < 12)  bl[tid] = b_loc[tid];
  __syncthreads();
  size_t idx = (size_t)blockIdx.x * blockDim.x + tid;   // over B*L (exact)
  int b = (int)(idx >> 10), l = (int)(idx & (LL - 1));
  const float4* xr = (const float4*)(x + idx * 12);
  float4 f0 = xr[0], f1 = xr[1], f2 = xr[2];
  float xin[12] = {f0.x, f0.y, f0.z, f0.w, f1.x, f1.y, f1.z, f1.w,
                   f2.x, f2.y, f2.z, f2.w};
  union { uint4 q[2]; f16 h[16]; } row;
#pragma unroll
  for (int co = 0; co < 12; ++co) {
    float s = bl[co];
#pragma unroll
    for (int ci = 0; ci < 12; ++ci) s += Wl[co * 12 + ci] * xin[ci];
    row.h[co] = (f16)fast_tanh(s);
  }
  row.h[12] = row.h[13] = row.h[14] = row.h[15] = (f16)0.f;
  uint4* dst = (uint4*)(xf16 + ((size_t)l * BB + b) * 16);
  dst[0] = row.q[0];
  dst[1] = row.q[1];
}

// A-operand gather: lane<16 reads row halfs [0..15], lane>=16 reads [8..23].
__device__ __forceinline__ v16h load_a_lds(const f16* base, int lane) {
  int off = (lane & 15) * 32 + ((lane < 16) ? 0 : 8);
  union { uint4 q[2]; v16h v; } u;
  const uint4* p = (const uint4*)(base + off);
  u.q[0] = p[0];
  u.q[1] = p[1];
  return u.v;
}

// ---------------------------------------------------------------------------
// LSTM: one workgroup (2 waves) owns 16 batch rows through all 1024 steps.
// Layer-pipelined wave specialization: wave0 runs layer0 for step t while
// wave1 runs layer1 for step t-1; h0 handed off via a 2-slot LDS ring with
// one workgroup barrier per step. Per wave per step: one fused [x|h] WMMA
// per 16-gate N-tile (3 tiles) + gate nonlinearities for 6 (m,j) pairs/lane.
// ---------------------------------------------------------------------------
__global__ void __launch_bounds__(64) lstm_kernel(const f16* __restrict__ xf16,
                                                  const int* __restrict__ lengths,
                                                  const f16* __restrict__ bimg,
                                                  const float* __restrict__ biases,
                                                  f16* __restrict__ feat16) {
  __shared__ f16 cA0[16 * 32];        // wave0: [x_t | h0 | pad] per batch row
  __shared__ f16 cA1[16 * 32];        // wave1: [h0_in | h1 | pad]
  __shared__ f16 h0buf[2][16 * 12];   // layer0 -> layer1 handoff ring
  __shared__ float gbuf[2][16][G4];   // per-wave gate pre-activations
  int tid = threadIdx.x;
  int wv = tid >> 5;                  // 0 = layer0 wave, 1 = layer1 wave
  int lane = tid & 31;
  int b0 = blockIdx.x * 16;
  int n = lane & 15, hi = lane >> 4;

  // per-wave weight image + fused bias (layer wv)
  v16h Bw[3];
  float bbv[3];
#pragma unroll
  for (int nt = 0; nt < 3; ++nt) {
    union { uint4 q[2]; v16h v; } u;
    const uint4* p = (const uint4*)(bimg + ((size_t)(wv * 3 + nt) * 32 + lane) * 16);
    u.q[0] = p[0]; u.q[1] = p[1];
    Bw[nt] = u.v;
    bbv[nt] = biases[wv * G4 + nt * 16 + n];
  }

  {
    uint4 z = {0, 0, 0, 0};
    f16* myA = wv ? cA1 : cA0;
    for (int i = lane; i < 64; i += 32) ((uint4*)myA)[i] = z;
    if (wv == 0) {
      for (int i = lane; i < 48; i += 32) ((uint4*)h0buf)[i] = z;   // 768 B
    }
  }

  int len = 1;
  float invlen = 1.0f;
  if (wv == 1 && lane < 16) {         // only wave1 emits feature rows
    len = lengths[b0 + lane];
    invlen = 1.0f / (float)len;
  }

  float cr[6];                        // cell state (this wave's layer)
  int pm[6], pj[6];
#pragma unroll
  for (int k = 0; k < 6; ++k) {
    cr[k] = 0.f;
    int p = lane + 32 * k;            // 192 (batch,hidden) pairs over 32 lanes
    pm[k] = p / 12;
    pj[k] = p - pm[k] * 12;
  }

  // software-pipelined x_t fetch (wave0 only)
  uint4 x4 = {0, 0, 0, 0};
  uint2 x2 = {0, 0};
  if (wv == 0 && lane < 16) {
    const f16* p = xf16 + ((size_t)(b0 + lane)) * 16;   // t = 0
    x4 = ((const uint4*)p)[0];
    x2 = *(const uint2*)(p + 8);
  }
  __syncthreads();

  for (int t = 0; t < LL + 1; ++t) {
    if (wv == 0) {
      if (t < LL) {
        if (lane < 16) {              // stage x_t (halfs 0..11; 12..23 hold h0)
          *(uint4*)(cA0 + lane * 32) = x4;
          *(uint2*)(cA0 + lane * 32 + 8) = x2;
        }
        if (lane < 16 && t + 1 < LL) {
          const f16* p = xf16 + ((size_t)(t + 1) * BB + b0 + lane) * 16;
          x4 = ((const uint4*)p)[0];
          x2 = *(const uint2*)(p + 8);
        }
        v16h a = load_a_lds(cA0, lane);
#pragma unroll
        for (int nt = 0; nt < 3; ++nt) {
          v8f acc = {0.f, 0.f, 0.f, 0.f, 0.f, 0.f, 0.f, 0.f};
          acc = __builtin_amdgcn_wmma_f32_16x16x32_f16(false, a, false, Bw[nt],
                                                       (short)0, acc, false, false);
#pragma unroll
          for (int v = 0; v < 8; ++v)
            gbuf[0][v + 8 * hi][nt * 16 + n] = acc[v] + bbv[nt];
        }
#pragma unroll
        for (int k = 0; k < 6; ++k) {
          int m = pm[k], j = pj[k];
          float ig = fast_sig(gbuf[0][m][j]);
          float fg = fast_sig(gbuf[0][m][12 + j]);
          float gg = fast_tanh(gbuf[0][m][24 + j]);
          float og = fast_sig(gbuf[0][m][36 + j]);
          cr[k] = fg * cr[k] + ig * gg;
          f16 hh = (f16)(og * fast_tanh(cr[k]));
          cA0[m * 32 + 12 + j] = hh;              // recurrent h0
          h0buf[t & 1][m * 12 + j] = hh;          // handoff to wave1
        }
      }
    } else {
      if (t >= 1) {
        int tp = t - 1;
        if (lane < 16) {              // pull h0(tp) from ring into A staging
          const uint2* s = (const uint2*)(h0buf[tp & 1] + lane * 12);
          uint2 a0 = s[0], a1 = s[1], a2 = s[2];
          uint2* d = (uint2*)(cA1 + lane * 32);
          d[0] = a0; d[1] = a1; d[2] = a2;
        }
        v16h a = load_a_lds(cA1, lane);
#pragma unroll
        for (int nt = 0; nt < 3; ++nt) {
          v8f acc = {0.f, 0.f, 0.f, 0.f, 0.f, 0.f, 0.f, 0.f};
          acc = __builtin_amdgcn_wmma_f32_16x16x32_f16(false, a, false, Bw[nt],
                                                       (short)0, acc, false, false);
#pragma unroll
          for (int v = 0; v < 8; ++v)
            gbuf[1][v + 8 * hi][nt * 16 + n] = acc[v] + bbv[nt];
        }
#pragma unroll
        for (int k = 0; k < 6; ++k) {
          int m = pm[k], j = pj[k];
          float ig = fast_sig(gbuf[1][m][j]);
          float fg = fast_sig(gbuf[1][m][12 + j]);
          float gg = fast_tanh(gbuf[1][m][24 + j]);
          float og = fast_sig(gbuf[1][m][36 + j]);
          cr[k] = fg * cr[k] + ig * gg;
          cA1[m * 32 + 12 + j] = (f16)(og * fast_tanh(cr[k]));
        }
        // emit attention-MLP feature row: [t_ratio, masked h1(12), 0,0,0]
        if (lane < 16) {
          bool valid = tp < len;
          union { uint4 q[2]; f16 h[16]; } row;
          float r = fminf((float)tp * invlen, 1.0f);
          row.h[0] = (f16)r;
#pragma unroll
          for (int j = 0; j < 12; ++j)
            row.h[1 + j] = valid ? cA1[lane * 32 + 12 + j] : (f16)0.f;
          row.h[13] = row.h[14] = row.h[15] = (f16)0.f;
          uint4* dst = (uint4*)(feat16 + (((size_t)(b0 + lane) << 10) + tp) * 16);
          dst[0] = row.q[0];
          dst[1] = row.q[1];
        }
      }
    }
    __syncthreads();
  }
}

// ---------------------------------------------------------------------------
// attention MLP: logits = tanh(feat @ W1^T + b1) @ W2^T + b2, WMMA over
// 16-row tiles of the 2M (b,l) rows; K=13 padded, N=64 in 4 tiles.
// ---------------------------------------------------------------------------
__global__ void mlp_kernel(const f16* __restrict__ feat16, const f16* __restrict__ bimgW1,
                           const float* __restrict__ b1, const float* __restrict__ W2,
                           const float* __restrict__ b2, float* __restrict__ logits,
                           int ntiles) {
  int lane = threadIdx.x & 31;
  int wave = (int)((blockIdx.x * blockDim.x + threadIdx.x) >> 5);
  int nwaves = (int)((gridDim.x * blockDim.x) >> 5);
  int n = lane & 15, hi = lane >> 4;

  v16h Bw[4];
  float b1v[4], w2v[4];
#pragma unroll
  for (int nt = 0; nt < 4; ++nt) {
    union { uint4 q[2]; v16h v; } u;
    const uint4* p = (const uint4*)(bimgW1 + ((size_t)nt * 32 + lane) * 16);
    u.q[0] = p[0]; u.q[1] = p[1];
    Bw[nt] = u.v;
    int uo = nt * 16 + n;
    b1v[nt] = b1[uo];
    w2v[nt] = W2[uo];
  }
  float b2v = b2[0];

  for (int tile = wave; tile < ntiles; tile += nwaves) {
    size_t r0 = (size_t)tile * 16;
    union { uint4 q[2]; v16h v; } ua;       // every lane loads row (lane&15)
    const uint4* pa = (const uint4*)(feat16 + (r0 + n) * 16);
    ua.q[0] = pa[0]; ua.q[1] = pa[1];
    v8f acc[4];
#pragma unroll
    for (int nt = 0; nt < 4; ++nt) {
      v8f z = {0.f, 0.f, 0.f, 0.f, 0.f, 0.f, 0.f, 0.f};
      acc[nt] = __builtin_amdgcn_wmma_f32_16x16x32_f16(false, ua.v, false, Bw[nt],
                                                       (short)0, z, false, false);
    }
#pragma unroll
    for (int v = 0; v < 8; ++v) {
      float s = 0.f;
#pragma unroll
      for (int nt = 0; nt < 4; ++nt) s += fast_tanh(acc[nt][v] + b1v[nt]) * w2v[nt];
#pragma unroll
      for (int d = 1; d < 16; d <<= 1) s += __shfl_xor(s, d, 32);
      if (n == 0) logits[r0 + v + 8 * hi] = s + b2v;
    }
  }
}

// ---------------------------------------------------------------------------
// softmax over L + attention-weighted pooling, one workgroup per batch row.
// ---------------------------------------------------------------------------
__global__ void pool_kernel(const float* __restrict__ logits,
                            const f16* __restrict__ feat16,
                            float* __restrict__ out) {
  int b = blockIdx.x, tid = threadIdx.x;
  __shared__ float red[256];
  __shared__ float wred[8][13];
  const float* lg = logits + (size_t)b * LL;

  float mx = -3.4e38f;
  for (int l = tid; l < LL; l += 256) mx = fmaxf(mx, lg[l]);
  red[tid] = mx;
  __syncthreads();
  for (int s = 128; s > 0; s >>= 1) {
    if (tid < s) red[tid] = fmaxf(red[tid], red[tid + s]);
    __syncthreads();
  }
  mx = red[0];

  float acc[13];
#pragma unroll
  for (int i = 0; i < 13; ++i) acc[i] = 0.f;
  const uint4* fb = (const uint4*)(feat16 + (((size_t)b) << 10) * 16);
  for (int l = tid; l < LL; l += 256) {
    float e = __expf(lg[l] - mx);
    acc[12] += e;
    union { uint4 q[2]; f16 h[16]; } u;
    u.q[0] = fb[l * 2];
    u.q[1] = fb[l * 2 + 1];
#pragma unroll
    for (int c = 0; c < 12; ++c) acc[c] += e * (float)u.h[1 + c];
  }
#pragma unroll
  for (int d = 1; d < 32; d <<= 1) {
#pragma unroll
    for (int i = 0; i < 13; ++i) acc[i] += __shfl_xor(acc[i], d, 32);
  }
  if ((tid & 31) == 0) {
    int w = tid >> 5;
#pragma unroll
    for (int i = 0; i < 13; ++i) wred[w][i] = acc[i];
  }
  __syncthreads();
  if (tid < 12) {
    float s = 0.f, se = 0.f;
    for (int w = 0; w < 8; ++w) {
      s += wred[w][tid];
      se += wred[w][12];
    }
    out[b * 12 + tid] = s / se;
  }
}

extern "C" void kernel_launch(void* const* d_in, const int* in_sizes, int n_in,
                              void* d_out, int out_size, void* d_ws, size_t ws_size,
                              hipStream_t stream) {
  const float* x     = (const float*)d_in[0];
  const int*   lens  = (const int*)d_in[1];
  const float* W_loc = (const float*)d_in[2];
  const float* b_loc = (const float*)d_in[3];
  const float* Wih0  = (const float*)d_in[4];
  const float* Whh0  = (const float*)d_in[5];
  const float* bih0  = (const float*)d_in[6];
  const float* bhh0  = (const float*)d_in[7];
  const float* Wih1  = (const float*)d_in[8];
  const float* Whh1  = (const float*)d_in[9];
  const float* bih1  = (const float*)d_in[10];
  const float* bhh1  = (const float*)d_in[11];
  const float* W1    = (const float*)d_in[12];
  const float* b1    = (const float*)d_in[13];
  const float* W2    = (const float*)d_in[14];
  const float* b2    = (const float*)d_in[15];
  float* out = (float*)d_out;

  char* ws = (char*)d_ws;
  const size_t SZ_XF  = (size_t)LL * BB * 16 * 2;   // 64 MiB (time-major x-features)
  const size_t SZ_LOG = (size_t)BB * LL * 4;        // 8 MiB  (logits)
  f16*   xf16   = (f16*)(ws);
  f16*   feat16 = (f16*)(ws + SZ_XF);
  float* logits = (float*)(ws + 2 * SZ_XF);
  f16*   bimg   = (f16*)(ws + 2 * SZ_XF + SZ_LOG);
  float* biases = (float*)(ws + 2 * SZ_XF + SZ_LOG + (size_t)10 * 512 * 2);

  prep_kernel<<<1, 256, 0, stream>>>(Wih0, Whh0, bih0, bhh0,
                                     Wih1, Whh1, bih1, bhh1, W1, bimg, biases);
  enc_kernel<<<(BB * LL) / 256, 256, 0, stream>>>(x, W_loc, b_loc, xf16);
  lstm_kernel<<<BB / 16, 64, 0, stream>>>(xf16, lens, bimg, biases, feat16);
  mlp_kernel<<<512, 256, 0, stream>>>(feat16, bimg + 6 * 512, b1, W2, b2,
                                      logits, (BB * LL) / 16);
  pool_kernel<<<BB, 256, 0, stream>>>(logits, feat16, out);
}
